// HypEqualizedLinear_27659589386533
// MI455X (gfx1250) — compile-verified, matrix operand order checked by reading, not run on hardware
//
#include <hip/hip_runtime.h>

typedef __bf16 bf16_t;
typedef __attribute__((ext_vector_type(16))) __bf16 v16bf;
typedef __attribute__((ext_vector_type(8)))  float  v8f;

#define BATCH 8192
#define DIN   2048
#define DOUT  2048

// ---------------- staging: x -> bf16, plus row norm ||x|| ----------------
__global__ __launch_bounds__(256) void k_conv_x(const float* __restrict__ x,
                                                bf16_t* __restrict__ xb,
                                                float* __restrict__ xnorm) {
  const int r = blockIdx.x;
  const float* xr = x + (size_t)r * DIN;
  bf16_t* xbr = xb + (size_t)r * DIN;
  float s = 0.f;
  #pragma unroll
  for (int i = 0; i < DIN / 256; ++i) {
    int c = threadIdx.x + i * 256;
    float v = xr[c];
    s += v * v;
    xbr[c] = (bf16_t)v;
  }
  __shared__ float red[8];
  int lane = threadIdx.x & 31, wid = threadIdx.x >> 5;
  #pragma unroll
  for (int o = 16; o > 0; o >>= 1) s += __shfl_down(s, o, 32);
  if (lane == 0) red[wid] = s;
  __syncthreads();
  if (wid == 0) {
    float t = (lane < 8) ? red[lane] : 0.f;
    #pragma unroll
    for (int o = 4; o > 0; o >>= 1) t += __shfl_down(t, o, 32);
    if (lane == 0) xnorm[r] = fmaxf(sqrtf(t), 1e-15f);
  }
}

// ---------------- staging: 0.1*W -> bf16 ----------------
__global__ __launch_bounds__(256) void k_conv_w(const float* __restrict__ w,
                                                bf16_t* __restrict__ wb) {
  size_t i = (size_t)blockIdx.x * 256 + threadIdx.x;
  wb[i] = (bf16_t)(w[i] * 0.1f);
}

// ---------------- b = expmap0(0.1*bias), y2 = ||b||^2 ----------------
__global__ __launch_bounds__(256) void k_bias(const float* __restrict__ bias,
                                              float* __restrict__ bh,
                                              float* __restrict__ y2) {
  __shared__ float red[8];
  __shared__ float tot_s;
  float u[DOUT / 256];
  float s = 0.f;
  #pragma unroll
  for (int i = 0; i < DOUT / 256; ++i) {
    u[i] = bias[threadIdx.x + i * 256] * 0.1f;
    s += u[i] * u[i];
  }
  int lane = threadIdx.x & 31, wid = threadIdx.x >> 5;
  #pragma unroll
  for (int o = 16; o > 0; o >>= 1) s += __shfl_down(s, o, 32);
  if (lane == 0) red[wid] = s;
  __syncthreads();
  if (threadIdx.x == 0) {
    float t = 0.f;
    #pragma unroll
    for (int w = 0; w < 8; ++w) t += red[w];
    tot_s = t;
  }
  __syncthreads();
  float tot = tot_s;
  float un = fmaxf(sqrtf(tot), 1e-15f);
  float sc = tanhf(un) / un;
  #pragma unroll
  for (int i = 0; i < DOUT / 256; ++i)
    bh[threadIdx.x + i * 256] = sc * u[i];
  if (threadIdx.x == 0) *y2 = sc * sc * tot;
}

// ---------------- main GEMM: mx = Xb @ Wb^T (bf16 WMMA, f32 acc) ----------------
// Block: 128x128, 8 waves as 2(M) x 4(N); wave tile 64x32 = 4x2 WMMA 16x16 tiles.
// Software-pipelined K loop (double-buffered fragments), single back-edge with
// peeled prologue/epilogue so the allocator keeps accumulators in place.
__global__ __launch_bounds__(256) void k_gemm(const bf16_t* __restrict__ A,
                                              const bf16_t* __restrict__ B,
                                              float* __restrict__ C) {
  const int lane = threadIdx.x & 31;
  const int wave = threadIdx.x >> 5;
  const int wm = wave & 1;    // 0..1  -> 64-row slab
  const int wn = wave >> 1;   // 0..3  -> 32-col slab
  const int bM = blockIdx.y * 128;
  const int bN = blockIdx.x * 128;
  const int r16 = lane & 15;
  const int kh = (lane >> 4) * 16;  // lanes 16..31 take the upper K-half

  v8f acc[4][2] = {};

  const bf16_t* Ap = A + (size_t)(bM + wm * 64 + r16) * DIN + kh;
  const bf16_t* Bp = B + (size_t)(bN + wn * 32 + r16) * DIN + kh;

  v16bf a0[4], b0[2], a1[4], b1[2];

  // prologue: load chunk 0
  #pragma unroll
  for (int i = 0; i < 4; ++i) a0[i] = *(const v16bf*)(Ap + (size_t)i * 16 * DIN);
  #pragma unroll
  for (int j = 0; j < 2; ++j) b0[j] = *(const v16bf*)(Bp + (size_t)j * 16 * DIN);

  // steady state: process chunk pairs (k0, k0+32); buf0 enters holding chunk k0
  #pragma unroll 1
  for (int it = 0; it < (DIN / 64) - 1; ++it) {
    const int k0 = it * 64;
    #pragma unroll
    for (int i = 0; i < 4; ++i)
      a1[i] = *(const v16bf*)(Ap + (size_t)i * 16 * DIN + (k0 + 32));
    #pragma unroll
    for (int j = 0; j < 2; ++j)
      b1[j] = *(const v16bf*)(Bp + (size_t)j * 16 * DIN + (k0 + 32));

    #pragma unroll
    for (int i = 0; i < 4; ++i)
      #pragma unroll
      for (int j = 0; j < 2; ++j)
        acc[i][j] = __builtin_amdgcn_wmma_f32_16x16x32_bf16(
            false, a0[i], false, b0[j], (short)0, acc[i][j], false, false);

    #pragma unroll
    for (int i = 0; i < 4; ++i)
      a0[i] = *(const v16bf*)(Ap + (size_t)i * 16 * DIN + (k0 + 64));
    #pragma unroll
    for (int j = 0; j < 2; ++j)
      b0[j] = *(const v16bf*)(Bp + (size_t)j * 16 * DIN + (k0 + 64));

    #pragma unroll
    for (int i = 0; i < 4; ++i)
      #pragma unroll
      for (int j = 0; j < 2; ++j)
        acc[i][j] = __builtin_amdgcn_wmma_f32_16x16x32_bf16(
            false, a1[i], false, b1[j], (short)0, acc[i][j], false, false);
  }

  // epilogue: buf0 holds chunk DIN-64; load final chunk DIN-32 and finish
  #pragma unroll
  for (int i = 0; i < 4; ++i)
    a1[i] = *(const v16bf*)(Ap + (size_t)i * 16 * DIN + (DIN - 32));
  #pragma unroll
  for (int j = 0; j < 2; ++j)
    b1[j] = *(const v16bf*)(Bp + (size_t)j * 16 * DIN + (DIN - 32));

  #pragma unroll
  for (int i = 0; i < 4; ++i)
    #pragma unroll
    for (int j = 0; j < 2; ++j)
      acc[i][j] = __builtin_amdgcn_wmma_f32_16x16x32_bf16(
          false, a0[i], false, b0[j], (short)0, acc[i][j], false, false);
  #pragma unroll
  for (int i = 0; i < 4; ++i)
    #pragma unroll
    for (int j = 0; j < 2; ++j)
      acc[i][j] = __builtin_amdgcn_wmma_f32_16x16x32_bf16(
          false, a1[i], false, b1[j], (short)0, acc[i][j], false, false);

  // D layout: VGPR v -> M = v + 8*(lane>=16), N = lane&15
  const int col0 = bN + wn * 32 + r16;
  const int rhi = 8 * (lane >> 4);
  #pragma unroll
  for (int i = 0; i < 4; ++i) {
    #pragma unroll
    for (int j = 0; j < 2; ++j) {
      int col = col0 + j * 16;
      int rowbase = bM + wm * 64 + i * 16 + rhi;
      #pragma unroll
      for (int v = 0; v < 8; ++v)
        C[(size_t)(rowbase + v) * DOUT + col] = acc[i][j][v];
    }
  }
}

// ---------------- row-wise hyperbolic epilogue ----------------
__global__ __launch_bounds__(256) void k_epilogue(const float* __restrict__ mx,
                                                  const float* __restrict__ xnorm,
                                                  const float* __restrict__ bh,
                                                  const float* __restrict__ y2p,
                                                  float* __restrict__ out) {
  const int r = blockIdx.x;
  const float* m = mx + (size_t)r * DOUT;
  float mv[DOUT / 256], bv[DOUT / 256];
  float s1 = 0.f, s2 = 0.f;
  #pragma unroll
  for (int i = 0; i < DOUT / 256; ++i) {
    int c = threadIdx.x + i * 256;
    mv[i] = m[c];
    bv[i] = bh[c];
    s1 += mv[i] * mv[i];
    s2 += mv[i] * bv[i];
  }
  __shared__ float rs1[8], rs2[8];
  int lane = threadIdx.x & 31, wid = threadIdx.x >> 5;
  #pragma unroll
  for (int o = 16; o > 0; o >>= 1) {
    s1 += __shfl_down(s1, o, 32);
    s2 += __shfl_down(s2, o, 32);
  }
  if (lane == 0) { rs1[wid] = s1; rs2[wid] = s2; }
  __syncthreads();
  float S1 = 0.f, S2 = 0.f;
  #pragma unroll
  for (int w = 0; w < 8; ++w) { S1 += rs1[w]; S2 += rs2[w]; }

  float xn = xnorm[r];        // already >= 1e-15
  float y2 = *y2p;
  float mxn = fmaxf(sqrtf(S1), 1e-15f);
  // artanh with the reference clip
  float xc = fminf(fmaxf(xn, -1.f + 1e-5f), 1.f - 1e-5f);
  float art = 0.5f * (log1pf(xc) - log1pf(-xc));
  float s = tanhf(mxn / xn * art) / mxn;       // res = s * mx
  float x2 = s * s * S1;
  float xy = s * S2;
  float alpha = 1.f + 2.f * xy + y2;
  float beta  = 1.f - x2;
  float gamma = fmaxf(1.f + 2.f * xy + x2 * y2, 1e-15f);
  float p = alpha * s / gamma;
  float q = beta / gamma;
  // ||h||^2 analytically from scalars
  float hn2 = (alpha * alpha * x2 + 2.f * alpha * beta * xy + beta * beta * y2)
              / (gamma * gamma);
  float hn = fmaxf(sqrtf(hn2), 1e-15f);
  const float maxn = 1.0f - 1e-3f;
  float sc = (hn > maxn) ? (maxn / hn) : 1.0f;

  float* o = out + (size_t)r * DOUT;
  #pragma unroll
  for (int i = 0; i < DOUT / 256; ++i) {
    int c = threadIdx.x + i * 256;
    float v = sc * (p * mv[i] + q * bv[i]);
    o[c] = (v >= 0.f) ? v : 0.2f * v;
  }
}

extern "C" void kernel_launch(void* const* d_in, const int* in_sizes, int n_in,
                              void* d_out, int out_size, void* d_ws, size_t ws_size,
                              hipStream_t stream) {
  const float* x    = (const float*)d_in[0];
  const float* w    = (const float*)d_in[1];
  const float* bias = (const float*)d_in[2];
  float* out = (float*)d_out;

  char* ws = (char*)d_ws;
  const size_t off_Xb = 0;
  const size_t off_Wb = off_Xb + (size_t)BATCH * DIN * sizeof(bf16_t);
  const size_t off_mx = off_Wb + (size_t)DOUT * DIN * sizeof(bf16_t);
  const size_t off_xn = off_mx + (size_t)BATCH * DOUT * sizeof(float);
  const size_t off_bh = off_xn + (size_t)BATCH * sizeof(float);
  const size_t off_y2 = off_bh + (size_t)DOUT * sizeof(float);

  bf16_t* Xb = (bf16_t*)(ws + off_Xb);
  bf16_t* Wb = (bf16_t*)(ws + off_Wb);
  float*  mxp = (float*)(ws + off_mx);
  float*  xn = (float*)(ws + off_xn);
  float*  bh = (float*)(ws + off_bh);
  float*  y2 = (float*)(ws + off_y2);

  k_conv_x<<<BATCH, 256, 0, stream>>>(x, Xb, xn);
  k_conv_w<<<(DOUT * DIN) / 256, 256, 0, stream>>>(w, Wb);
  k_bias<<<1, 256, 0, stream>>>(bias, bh, y2);
  dim3 g(DOUT / 128, BATCH / 128);
  k_gemm<<<g, 256, 0, stream>>>(Xb, Wb, mxp);
  k_epilogue<<<BATCH, 256, 0, stream>>>(mxp, xn, bh, y2, out);
}
